// MalwareGNN_46385646797176
// MI455X (gfx1250) — compile-verified, matrix operand order checked by reading, not run on hardware
//
#include <hip/hip_runtime.h>
#include <hip/hip_bf16.h>

typedef __attribute__((ext_vector_type(16))) __bf16 v16bf;
typedef __attribute__((ext_vector_type(8)))  __bf16 v8bf;
typedef __attribute__((ext_vector_type(8)))  float  v8f;

#define GNN_N 100000
#define GNN_E 1600000
#define GNN_B 64
#define GNN_EPS 1e-5f

// ---------------------------------------------------------------- utilities
__global__ void zero_f32(float* p, int n) {
    int i = blockIdx.x * blockDim.x + threadIdx.x;
    if (i < n) p[i] = 0.0f;
}

// ---------------------------------------------------------------- degree / dinv
__global__ void degree_kernel(const int* __restrict__ dst, int E, float* __restrict__ deg) {
    int i = blockIdx.x * blockDim.x + threadIdx.x;
    if (i < E) atomicAdd(&deg[dst[i]], 1.0f);
}

__global__ void dinv_kernel(float* deg, int n) {
    int i = blockIdx.x * blockDim.x + threadIdx.x;
    if (i < n) deg[i] = rsqrtf(deg[i] + 1.0f);   // +1 = self loop
}

// ---------------------------------------------------------------- edge aggregation
// One thread per (edge, 4-feature chunk): consecutive lanes -> consecutive
// features of the same edge => coalesced gather loads and scatter atomics.
__global__ void agg_kernel(const float* __restrict__ x, const int* __restrict__ src,
                           const int* __restrict__ dst, const float* __restrict__ dinv,
                           float* __restrict__ out, int E, int n, int F, int cshift) {
    long long idx = (long long)blockIdx.x * blockDim.x + threadIdx.x;
    int chunks = 1 << cshift;                       // F/4
    long long total = (long long)(E + n) << cshift;
    if (idx >= total) return;
    int e  = (int)(idx >> cshift);
    int fc = (int)(idx & (chunks - 1));
    int s, d;
    if (e < E) { s = src[e]; d = dst[e]; }
    else       { s = e - E; d = s; }
    float norm = dinv[s] * dinv[d];
    int f = fc << 2;
    float4 v = *(const float4*)(x + (size_t)s * F + f);
    float* od = out + (size_t)d * F + f;
    atomicAdd(&od[0], v.x * norm);
    atomicAdd(&od[1], v.y * norm);
    atomicAdd(&od[2], v.z * norm);
    atomicAdd(&od[3], v.w * norm);
}

// ---------------------------------------------------------------- weight pre-pack
// Wt[m*K + k] = bf16(W[k*M + m]) : transposed bf16 copy so each WMMA B-fragment
// half is 8 contiguous bf16 (one 16B load).
__global__ void pack_wT_bf16(const float* __restrict__ W, __bf16* __restrict__ Wt,
                             int K, int M) {
    int i = blockIdx.x * blockDim.x + threadIdx.x;
    if (i >= K * M) return;
    int m = i / K, k = i % K;
    Wt[i] = (__bf16)W[(size_t)k * M + m];
}

// ---------------------------------------------------------------- bf16 WMMA GEMM
// Y[n,M] = X[n,K] @ W[K,M] + bias.  n%16==0, K%32==0, M%64==0.
// One wave per 16x64 output strip: one A fragment feeds 4 WMMAs per k-step.
// B operand from pre-packed transposed bf16 (contiguous 16B loads).
#define GEMM_TILE(ACC, TOFF)                                                          \
    {                                                                                 \
        const __bf16* wc = Wt + (size_t)(col0 + (TOFF) + r) * K + k0 + koff;          \
        v8bf blo = *(const v8bf*)(wc);                                                \
        v8bf bhi = *(const v8bf*)(wc + 16);                                           \
        v16bf b = __builtin_shufflevector(blo, bhi, 0, 1, 2, 3, 4, 5, 6, 7,           \
                                          8, 9, 10, 11, 12, 13, 14, 15);              \
        ACC = __builtin_amdgcn_wmma_f32_16x16x32_bf16(false, a, false, b,             \
                                                      (short)0, ACC, false, false);   \
    }

#define GEMM_STORE(ACC, TOFF)                                                         \
    {                                                                                 \
        int col = col0 + (TOFF) + (lane & 15);                                        \
        float bv = bias[col];                                                         \
        _Pragma("unroll")                                                             \
        for (int i = 0; i < 8; ++i)                                                   \
            Y[(size_t)(rbase + i) * M + col] = ACC[i] + bv;                           \
    }

__global__ void gemm_bias_wmma(const float* __restrict__ X, const __bf16* __restrict__ Wt,
                               const float* __restrict__ bias, float* __restrict__ Y,
                               int n, int K, int M) {
    int wave = threadIdx.x >> 5;
    int lane = threadIdx.x & 31;
    int wid = blockIdx.x * (blockDim.x >> 5) + wave;
    int mt = M >> 6;                      // 64-wide strips
    int nt = n >> 4;
    if (wid >= nt * mt) return;           // wave-uniform exit: EXEC all-1 at WMMA
    int rt = wid / mt, ct = wid % mt;
    int row0 = rt << 4, col0 = ct << 6;

    int r = lane & 15;
    int koff = (lane >> 4) << 3;          // 0 for lanes 0-15, 8 for lanes 16-31

    v8f acc0 = {}, acc1 = {}, acc2 = {}, acc3 = {};
    for (int k0 = 0; k0 < K; k0 += 32) {
        // A fragment: row (row0+r), K = k0+koff..+7 and k0+16+koff..+7
        // (ISA 16-bit A-matrix 16x32 layout, wave32)
        const float* xp = X + (size_t)(row0 + r) * K + k0 + koff;
        __builtin_prefetch(xp + 32, 0, 1);          // global_prefetch next k-panel
        float4 a0 = *(const float4*)(xp);
        float4 a1 = *(const float4*)(xp + 4);
        float4 a2 = *(const float4*)(xp + 16);
        float4 a3 = *(const float4*)(xp + 20);
        v16bf a;
        a[0]  = (__bf16)a0.x; a[1]  = (__bf16)a0.y; a[2]  = (__bf16)a0.z; a[3]  = (__bf16)a0.w;
        a[4]  = (__bf16)a1.x; a[5]  = (__bf16)a1.y; a[6]  = (__bf16)a1.z; a[7]  = (__bf16)a1.w;
        a[8]  = (__bf16)a2.x; a[9]  = (__bf16)a2.y; a[10] = (__bf16)a2.z; a[11] = (__bf16)a2.w;
        a[12] = (__bf16)a3.x; a[13] = (__bf16)a3.y; a[14] = (__bf16)a3.z; a[15] = (__bf16)a3.w;

        GEMM_TILE(acc0, 0)
        GEMM_TILE(acc1, 16)
        GEMM_TILE(acc2, 32)
        GEMM_TILE(acc3, 48)
    }

    // C/D layout: VGPR i -> row = i + (lane<16 ? 0 : 8), col = lane&15
    int rbase = row0 + ((lane >> 4) << 3);
    GEMM_STORE(acc0, 0)
    GEMM_STORE(acc1, 16)
    GEMM_STORE(acc2, 32)
    GEMM_STORE(acc3, 48)
}

// ---------------------------------------------------------------- BatchNorm
__global__ void bn_partial(const float* __restrict__ x, int n, int F,
                           float* __restrict__ sum, float* __restrict__ sumsq) {
    int c = threadIdx.x;
    float s = 0.0f, ss = 0.0f;
    for (int i = blockIdx.x; i < n; i += gridDim.x) {
        float v = x[(size_t)i * F + c];
        s += v; ss += v * v;
    }
    atomicAdd(&sum[c], s);
    atomicAdd(&sumsq[c], ss);
}

__global__ void bn_finalize(const float* __restrict__ sum, const float* __restrict__ sumsq,
                            int n, float* __restrict__ mu, float* __restrict__ rstd) {
    int c = threadIdx.x;
    float m = sum[c] / (float)n;
    float var = sumsq[c] / (float)n - m * m;
    mu[c] = m;
    rstd[c] = rsqrtf(var + GNN_EPS);
}

__global__ void bn_apply_relu(float* x, int n, int F, const float* __restrict__ mu,
                              const float* __restrict__ rstd, const float* __restrict__ g,
                              const float* __restrict__ be) {
    int i = blockIdx.x * blockDim.x + threadIdx.x;
    if (i >= n * F) return;
    int c = i % F;
    float v = (x[i] - mu[c]) * rstd[c] * g[c] + be[c];
    x[i] = v > 0.0f ? v : 0.0f;
}

// ---------------------------------------------------------------- LayerNorm + attention logits
__global__ void ln_att(float* x, int n, const float* __restrict__ ln_g,
                       const float* __restrict__ ln_b, const float* __restrict__ att,
                       float* __restrict__ ew, float* __restrict__ ssum) {
    int i = blockIdx.x * blockDim.x + threadIdx.x;
    float e = 0.0f;
    if (i < n) {
        float* row = x + (size_t)i * 64;
        float m = 0.0f;
        #pragma unroll
        for (int f = 0; f < 64; ++f) m += row[f];
        m *= (1.0f / 64.0f);
        float var = 0.0f;
        #pragma unroll
        for (int f = 0; f < 64; ++f) { float d = row[f] - m; var += d * d; }
        var *= (1.0f / 64.0f);
        float rs = rsqrtf(var + GNN_EPS);
        float t = 0.0f;
        #pragma unroll
        for (int f = 0; f < 64; ++f) {
            float y = (row[f] - m) * rs * ln_g[f] + ln_b[f];
            row[f] = y;
            t += y * att[f];
        }
        e = expf(tanhf(t));                 // tanh in (-1,1): max-free softmax is safe
        ew[i] = e;
    }
    __shared__ float sh[256];
    sh[threadIdx.x] = e;
    __syncthreads();
    for (int o = 128; o > 0; o >>= 1) {
        if (threadIdx.x < o) sh[threadIdx.x] += sh[threadIdx.x + o];
        __syncthreads();
    }
    if (threadIdx.x == 0) atomicAdd(ssum, sh[0]);
}

// ---------------------------------------------------------------- weighted per-graph pooling
__global__ void pool_kernel(const float* __restrict__ x, const float* __restrict__ ew,
                            const float* __restrict__ ssum, const int* __restrict__ batch,
                            float* __restrict__ s, float* __restrict__ cnt, int n) {
    int idx = blockIdx.x * blockDim.x + threadIdx.x;
    if (idx >= n * 64) return;
    int i = idx >> 6;
    int f = idx & 63;
    int g = batch[i];
    float w = ew[i] / ssum[0];
    atomicAdd(&s[g * 64 + f], x[idx] * w);
    if (f == 0) atomicAdd(&cnt[g], 1.0f);
}

// ---------------------------------------------------------------- tiny MLP head
__global__ void mlp_kernel(const float* __restrict__ s, const float* __restrict__ cnt,
                           const float* __restrict__ M0W, const float* __restrict__ M0b,
                           const float* __restrict__ M1W, const float* __restrict__ M1b,
                           const float* __restrict__ M2W, const float* __restrict__ M2b,
                           float* __restrict__ out) {
    int g = threadIdx.x;
    if (g >= GNN_B) return;
    float c = cnt[g]; c = c > 1.0f ? c : 1.0f;
    float inv = 1.0f / c;
    float h1[32];
    for (int j = 0; j < 32; ++j) {
        float a = M0b[j];
        for (int k = 0; k < 64; ++k) a += s[g * 64 + k] * inv * M0W[k * 32 + j];
        h1[j] = a > 0.0f ? a : 0.0f;
    }
    float h2[16];
    for (int j = 0; j < 16; ++j) {
        float a = M1b[j];
        for (int k = 0; k < 32; ++k) a += h1[k] * M1W[k * 16 + j];
        h2[j] = a > 0.0f ? a : 0.0f;
    }
    for (int j = 0; j < 2; ++j) {
        float a = M2b[j];
        for (int k = 0; k < 16; ++k) a += h2[k] * M2W[k * 2 + j];
        out[g * 2 + j] = a;
    }
}

// ---------------------------------------------------------------- host orchestration
static inline int cdiv(long long a, int b) { return (int)((a + b - 1) / b); }

extern "C" void kernel_launch(void* const* d_in, const int* in_sizes, int n_in,
                              void* d_out, int out_size, void* d_ws, size_t ws_size,
                              hipStream_t stream) {
    const float* x    = (const float*)d_in[0];
    const int*   edge = (const int*)d_in[1];
    const int*   batch = (const int*)d_in[2];
    const float* W0 = (const float*)d_in[3];
    const float* b0 = (const float*)d_in[4];
    const float* g0 = (const float*)d_in[5];
    const float* be0 = (const float*)d_in[6];
    const float* W1 = (const float*)d_in[7];
    const float* b1 = (const float*)d_in[8];
    const float* g1 = (const float*)d_in[9];
    const float* be1 = (const float*)d_in[10];
    const float* W2 = (const float*)d_in[11];
    const float* b2 = (const float*)d_in[12];
    const float* g2 = (const float*)d_in[13];
    const float* be2 = (const float*)d_in[14];
    const float* ln_g = (const float*)d_in[15];
    const float* ln_b = (const float*)d_in[16];
    const float* att  = (const float*)d_in[17];
    const float* M0W = (const float*)d_in[18];
    const float* M0b = (const float*)d_in[19];
    const float* M1W = (const float*)d_in[20];
    const float* M1b = (const float*)d_in[21];
    const float* M2W = (const float*)d_in[22];
    const float* M2b = (const float*)d_in[23];
    float* out = (float*)d_out;

    const int n = GNN_N, E = GNN_E;
    const int* srcI = edge;         // edge_index[0]
    const int* dstI = edge + E;     // edge_index[1]

    float* ws = (float*)d_ws;
    float* dinv  = ws;                        // n
    float* mu    = ws + n;                    // 256
    float* rstd  = mu + 256;                  // 256
    float* bsum  = rstd + 256;                // 256
    float* bsq   = bsum + 256;                // 256
    float* ew    = bsq + 256;                 // n
    float* ssum  = ew + n;                    // 1
    float* spool = ssum + 1;                  // 64*64
    float* cnt   = spool + GNN_B * 64;        // 64
    __bf16* wt16 = (__bf16*)(ws + 2 * (size_t)n + 8192);   // 256*256 bf16 = 32768 floats
    size_t big0 = 2 * (size_t)n + 8192 + 32768;            // 16B-aligned float offset
    float* bufA = ws + big0;                  // n*256
    float* bufB = bufA + (size_t)n * 256;     // n*256

    const int TPB = 256;

    // --- degrees -> dinv
    zero_f32<<<cdiv(n, TPB), TPB, 0, stream>>>(dinv, n);
    degree_kernel<<<cdiv(E, TPB), TPB, 0, stream>>>(dstI, E, dinv);
    dinv_kernel<<<cdiv(n, TPB), TPB, 0, stream>>>(dinv, n);

    // --- layer 1: (A x) W0 + b0, BN, ReLU    (x:128 -> 256)
    zero_f32<<<cdiv((long long)n * 128, TPB), TPB, 0, stream>>>(bufA, n * 128);
    agg_kernel<<<cdiv((long long)(E + n) * 32, TPB), TPB, 0, stream>>>(
        x, srcI, dstI, dinv, bufA, E, n, 128, 5);
    pack_wT_bf16<<<cdiv(128 * 256, TPB), TPB, 0, stream>>>(W0, wt16, 128, 256);
    gemm_bias_wmma<<<cdiv((long long)(n / 16) * (256 / 64), 8), TPB, 0, stream>>>(
        bufA, wt16, b0, bufB, n, 128, 256);
    zero_f32<<<2, TPB, 0, stream>>>(bsum, 512);
    bn_partial<<<512, 256, 0, stream>>>(bufB, n, 256, bsum, bsq);
    bn_finalize<<<1, 256, 0, stream>>>(bsum, bsq, n, mu, rstd);
    bn_apply_relu<<<cdiv((long long)n * 256, TPB), TPB, 0, stream>>>(bufB, n, 256, mu, rstd, g0, be0);

    // --- layer 2: 256 -> 128
    zero_f32<<<cdiv((long long)n * 256, TPB), TPB, 0, stream>>>(bufA, n * 256);
    agg_kernel<<<cdiv((long long)(E + n) * 64, TPB), TPB, 0, stream>>>(
        bufB, srcI, dstI, dinv, bufA, E, n, 256, 6);
    pack_wT_bf16<<<cdiv(256 * 128, TPB), TPB, 0, stream>>>(W1, wt16, 256, 128);
    gemm_bias_wmma<<<cdiv((long long)(n / 16) * (128 / 64), 8), TPB, 0, stream>>>(
        bufA, wt16, b1, bufB, n, 256, 128);
    zero_f32<<<2, TPB, 0, stream>>>(bsum, 512);
    bn_partial<<<512, 128, 0, stream>>>(bufB, n, 128, bsum, bsq);
    bn_finalize<<<1, 128, 0, stream>>>(bsum, bsq, n, mu, rstd);
    bn_apply_relu<<<cdiv((long long)n * 128, TPB), TPB, 0, stream>>>(bufB, n, 128, mu, rstd, g1, be1);

    // --- layer 3: 128 -> 64
    zero_f32<<<cdiv((long long)n * 128, TPB), TPB, 0, stream>>>(bufA, n * 128);
    agg_kernel<<<cdiv((long long)(E + n) * 32, TPB), TPB, 0, stream>>>(
        bufB, srcI, dstI, dinv, bufA, E, n, 128, 5);
    pack_wT_bf16<<<cdiv(128 * 64, TPB), TPB, 0, stream>>>(W2, wt16, 128, 64);
    gemm_bias_wmma<<<cdiv((long long)(n / 16) * (64 / 64), 8), TPB, 0, stream>>>(
        bufA, wt16, b2, bufB, n, 128, 64);
    zero_f32<<<2, TPB, 0, stream>>>(bsum, 512);
    bn_partial<<<512, 64, 0, stream>>>(bufB, n, 64, bsum, bsq);
    bn_finalize<<<1, 64, 0, stream>>>(bsum, bsq, n, mu, rstd);
    bn_apply_relu<<<cdiv((long long)n * 64, TPB), TPB, 0, stream>>>(bufB, n, 64, mu, rstd, g2, be2);

    // --- LayerNorm + attention + softmax sum
    zero_f32<<<cdiv(1 + GNN_B * 64 + GNN_B, TPB), TPB, 0, stream>>>(ssum, 1 + GNN_B * 64 + GNN_B);
    ln_att<<<cdiv(n, TPB), TPB, 0, stream>>>(bufB, n, ln_g, ln_b, att, ew, ssum);

    // --- weighted pooling per graph
    pool_kernel<<<cdiv((long long)n * 64, TPB), TPB, 0, stream>>>(bufB, ew, ssum, batch, spool, cnt, n);

    // --- MLP head
    mlp_kernel<<<1, 64, 0, stream>>>(spool, cnt, M0W, M0b, M1W, M1b, M2W, M2b, out);
}
